// GemmaBlock_30210799960139
// MI455X (gfx1250) — matrix operand order repaired and checked by reading, not compiled, not run on hardware
//
#include <hip/hip_runtime.h>

// ---------------------------------------------------------------------------
// GemmaBlock for MI455X (gfx1250): bf16 WMMA GEMMs with TDM (tensor_load_to_lds)
// double-buffered LDS staging.  B=2, T=2048, E=2048, H=8, KV=1, D=256, I=16384.
// ---------------------------------------------------------------------------

typedef __attribute__((ext_vector_type(16))) unsigned short v16us;
typedef __attribute__((ext_vector_type(8)))  unsigned short v8us;
typedef __attribute__((ext_vector_type(16))) __bf16         v16bf;
typedef __attribute__((ext_vector_type(8)))  float          v8f;
typedef __attribute__((ext_vector_type(4)))  unsigned int   u32x4;
typedef __attribute__((ext_vector_type(8)))  int            i32x8;
typedef __attribute__((ext_vector_type(4)))  int            i32x4;

static constexpr int GB_B = 2;
static constexpr int GB_T = 2048;
static constexpr int GB_E = 2048;
static constexpr int GB_H = 8;
static constexpr int GB_D = 256;
static constexpr int GB_I = 16384;
static constexpr float GB_EPS = 1e-6f;

__device__ __forceinline__ unsigned short f2bf(float f) {
    unsigned int u = __float_as_uint(f);
    u += 0x7fffu + ((u >> 16) & 1u);          // round-to-nearest-even
    return (unsigned short)(u >> 16);
}

// ---------------------------------------------------------------------------
// TDM: async 2D tile load (rows x 32 bf16 elements, row stride ld elements)
// from global memory into LDS at byte offset lds_off.  D# built per ISA
// CDNA5 ISA.  Tracked by TENSORcnt.
// ---------------------------------------------------------------------------
__device__ __forceinline__ void tdm_load_tile_2d(unsigned lds_off,
                                                 const void* gptr,
                                                 unsigned rows,
                                                 unsigned long long ld_elems) {
    const unsigned long long ga = (unsigned long long)gptr;
    u32x4 g0 = {};
    g0[0] = 1u;                                          // count=1, user mode
    g0[1] = lds_off;                                     // lds_addr (bytes)
    g0[2] = (unsigned)(ga & 0xffffffffull);              // global_addr[31:0]
    g0[3] = (unsigned)((ga >> 32) & 0x1ffffffull)        // global_addr[56:32]
          | (2u << 30);                                  // type=2 ("image")
    i32x8 g1 = {};
    g1[0] = (int)(1u << 16);                             // mask=0, data_size=1 (2B)
    g1[1] = (int)(32u << 16);                            // tensor_dim0[15:0]=32
    g1[2] = (int)((rows & 0xffffu) << 16);               // dim0 hi=0, tensor_dim1 lo
    g1[3] = (int)(32u << 16);                            // dim1 hi=0, tile_dim0=32
    g1[4] = (int)(rows & 0xffffu);                       // tile_dim1=rows, tile_dim2=0
    g1[5] = (int)(unsigned)(ld_elems & 0xffffffffull);   // tensor_dim0_stride[31:0]
    g1[6] = (int)(unsigned)((ld_elems >> 32) & 0xffffull); // stride[47:32]
    g1[7] = 0;
    const i32x4 z4 = {};
#if __clang_major__ >= 23
    const i32x8 z8 = {};
    __builtin_amdgcn_tensor_load_to_lds(g0, g1, z4, z4, z8, 0);
#else
    __builtin_amdgcn_tensor_load_to_lds(g0, g1, z4, z4, 0);
#endif
}

// ---------------------------------------------------------------------------
// fp32 [K,N] -> bf16 transposed [N,K] (LDS-tiled for coalescing)
// ---------------------------------------------------------------------------
__global__ __launch_bounds__(256) void transpose_f32_bf16(
    const float* __restrict__ W, unsigned short* __restrict__ Wt, int K, int N) {
    __shared__ float tile[32][33];
    const int n  = blockIdx.x * 32 + threadIdx.x;
    const int k0 = blockIdx.y * 32;
    #pragma unroll
    for (int j = threadIdx.y; j < 32; j += 8)
        tile[j][threadIdx.x] = W[(long)(k0 + j) * N + n];
    __syncthreads();
    const int k  = k0 + (int)threadIdx.x;
    const int nb = blockIdx.x * 32;
    #pragma unroll
    for (int j = threadIdx.y; j < 32; j += 8)
        Wt[(long)(nb + j) * K + k] = f2bf(tile[threadIdx.x][j]);
}

// ---------------------------------------------------------------------------
// RMSNorm row kernel: out_bf16 = x * rsqrt(mean(x^2)+eps) * (1+g)
// ---------------------------------------------------------------------------
__global__ __launch_bounds__(256) void rmsnorm_bf16(
    const float* __restrict__ X, const float* __restrict__ g,
    unsigned short* __restrict__ Out, int E) {
    const long row = blockIdx.x;
    const float* x = X + row * E;
    float vals[8];
    float ss = 0.f;
    #pragma unroll
    for (int i = 0; i < 8; ++i) {
        const int e = (int)threadIdx.x + i * 256;
        vals[i] = x[e];
        ss += vals[i] * vals[i];
    }
    #pragma unroll
    for (int o = 16; o > 0; o >>= 1) ss += __shfl_xor(ss, o, 32);
    __shared__ float red[8];
    if ((threadIdx.x & 31) == 0) red[threadIdx.x >> 5] = ss;
    __syncthreads();
    float tot = 0.f;
    #pragma unroll
    for (int i = 0; i < 8; ++i) tot += red[i];
    const float inv = rsqrtf(tot / (float)E + GB_EPS);
    #pragma unroll
    for (int i = 0; i < 8; ++i) {
        const int e = (int)threadIdx.x + i * 256;
        Out[row * E + e] = f2bf(vals[i] * inv * (1.f + g[e]));
    }
}

// ---------------------------------------------------------------------------
// Row softmax: scores f32 [rows, T] -> probs bf16. One block per row.
// ---------------------------------------------------------------------------
__global__ __launch_bounds__(256) void softmax_bf16(
    const float* __restrict__ S, unsigned short* __restrict__ P, int T) {
    const long row = blockIdx.x;
    const float* s = S + row * T;
    __shared__ float red[8];
    float v[8];
    float mx = -3.4e38f;
    #pragma unroll
    for (int i = 0; i < 8; ++i) {
        v[i] = s[(int)threadIdx.x + i * 256];
        mx = fmaxf(mx, v[i]);
    }
    #pragma unroll
    for (int o = 16; o > 0; o >>= 1) mx = fmaxf(mx, __shfl_xor(mx, o, 32));
    if ((threadIdx.x & 31) == 0) red[threadIdx.x >> 5] = mx;
    __syncthreads();
    float mall = red[0];
    #pragma unroll
    for (int i = 1; i < 8; ++i) mall = fmaxf(mall, red[i]);
    __syncthreads();
    float sum = 0.f;
    #pragma unroll
    for (int i = 0; i < 8; ++i) {
        v[i] = __expf(v[i] - mall);
        sum += v[i];
    }
    #pragma unroll
    for (int o = 16; o > 0; o >>= 1) sum += __shfl_xor(sum, o, 32);
    if ((threadIdx.x & 31) == 0) red[threadIdx.x >> 5] = sum;
    __syncthreads();
    float tot = 0.f;
    #pragma unroll
    for (int i = 0; i < 8; ++i) tot += red[i];
    const float inv = 1.f / tot;
    #pragma unroll
    for (int i = 0; i < 8; ++i)
        P[row * T + (int)threadIdx.x + i * 256] = f2bf(v[i] * inv);
}

// ---------------------------------------------------------------------------
// bf16 WMMA GEMM with TDM double-buffered LDS staging.
//   C[z] = A[z] (MxK) * Bt[z]^T   (Bt stored NxK row-major)
// Block = 256 thr = 8 waves; block tile 128(M) x 64(N); wave tile 16x64.
// Per K-step (32): TDM loads A-tile 128x32 and B-tile 64x32 into LDS;
// all 8 waves share the B tile from LDS.
// ---------------------------------------------------------------------------
enum { EPI_BF16 = 0, EPI_BF16_T = 1, EPI_F32_SCALE = 2,
       EPI_F32_BIAS_RES = 3, EPI_BF16_GELU = 4 };

__device__ __forceinline__ v16bf frag_from_lds_a(const unsigned short* sA,
                                                 int row, int laneHi) {
    // A 16-bit layout: lane group lo holds K {0..7,16..23}, hi {8..15,24..31}
    v8us a0 = *(const v8us*)(sA + row * 32 + laneHi * 8);
    v8us a1 = *(const v8us*)(sA + row * 32 + 16 + laneHi * 8);
    v16us av;
    #pragma unroll
    for (int i = 0; i < 8; ++i) { av[i] = a0[i]; av[8 + i] = a1[i]; }
    return __builtin_bit_cast(v16bf, av);
}

__device__ __forceinline__ v16bf frag_from_lds_b(const unsigned short* sB,
                                                 int col, int laneHi) {
    // B 16-bit layout: lane = column, lane group lo holds K {0..7,16..23}, hi {8..15,24..31}
    v8us b0 = *(const v8us*)(sB + col * 32 + laneHi * 8);
    v8us b1 = *(const v8us*)(sB + col * 32 + 16 + laneHi * 8);
    v16us bv;
    #pragma unroll
    for (int i = 0; i < 8; ++i) { bv[i] = b0[i]; bv[8 + i] = b1[i]; }
    return __builtin_bit_cast(v16bf, bv);
}

template <int EPI>
__global__ __launch_bounds__(256) void gemm_bf16_wmma(
    const unsigned short* __restrict__ A, int lda, long sAb, long sAh,
    const unsigned short* __restrict__ Bt, int ldb, long sBb, long sBh,
    void* __restrict__ OutV, int ldo, long sOb, long sOh,
    int zdiv,
    const float* __restrict__ bias,
    const float* __restrict__ resid, int ldr,
    int K, float scale) {

    __shared__ unsigned short ldsA[2][128 * 32];   // 16 KB
    __shared__ unsigned short ldsB[2][64 * 32];    //  8 KB

    const int lane   = threadIdx.x & 31;
    const int wave   = threadIdx.x >> 5;
    const int laneLo = lane & 15;
    const int laneHi = lane >> 4;

    const int z  = blockIdx.z;
    const int zb = z / zdiv;
    const int zh = z - zb * zdiv;
    A  += (long)zb * sAb + (long)zh * sAh;
    Bt += (long)zb * sBb + (long)zh * sBh;
    const long ob = (long)zb * sOb + (long)zh * sOh;

    const int m0 = blockIdx.y * 128 + wave * 16;
    const int n0 = blockIdx.x * 64;

    const unsigned short* aTile = A + (long)(blockIdx.y * 128) * lda;
    const unsigned short* bTile = Bt + (long)n0 * ldb;

    // LDS generic address carries the LDS byte offset in addr[31:0].
    const unsigned ldsA_off0 = (unsigned)(size_t)(void*)&ldsA[0][0];
    const unsigned ldsA_off1 = (unsigned)(size_t)(void*)&ldsA[1][0];
    const unsigned ldsB_off0 = (unsigned)(size_t)(void*)&ldsB[0][0];
    const unsigned ldsB_off1 = (unsigned)(size_t)(void*)&ldsB[1][0];

    const int nk = K >> 5;   // K-steps of 32

    if (wave == 0) {         // TDM issue is a wave-level op; one wave drives it
        tdm_load_tile_2d(ldsA_off0, aTile, 128, (unsigned long long)lda);
        tdm_load_tile_2d(ldsB_off0, bTile, 64, (unsigned long long)ldb);
    }

    v8f acc[4] = {v8f{}, v8f{}, v8f{}, v8f{}};

    for (int kt = 0; kt < nk; ++kt) {
        if (wave == 0) {
            if (kt + 1 < nk) {
                const unsigned aOff = ((kt + 1) & 1) ? ldsA_off1 : ldsA_off0;
                const unsigned bOff = ((kt + 1) & 1) ? ldsB_off1 : ldsB_off0;
                tdm_load_tile_2d(aOff, aTile + (long)(kt + 1) * 32, 128,
                                 (unsigned long long)lda);
                tdm_load_tile_2d(bOff, bTile + (long)(kt + 1) * 32, 64,
                                 (unsigned long long)ldb);
                __builtin_amdgcn_s_wait_tensorcnt(2);  // tile kt landed
            } else {
                __builtin_amdgcn_s_wait_tensorcnt(0);
            }
        }
        __syncthreads();

        const unsigned short* sA = &ldsA[kt & 1][0];
        const unsigned short* sB = &ldsB[kt & 1][0];

        const v16bf afr = frag_from_lds_a(sA, wave * 16 + laneLo, laneHi);
        const v16bf b0  = frag_from_lds_b(sB,  0 + laneLo, laneHi);
        const v16bf b1  = frag_from_lds_b(sB, 16 + laneLo, laneHi);
        const v16bf b2  = frag_from_lds_b(sB, 32 + laneLo, laneHi);
        const v16bf b3  = frag_from_lds_b(sB, 48 + laneLo, laneHi);

        acc[0] = __builtin_amdgcn_wmma_f32_16x16x32_bf16(false, afr, false, b0,
                                                         (short)0, acc[0], false, false);
        acc[1] = __builtin_amdgcn_wmma_f32_16x16x32_bf16(false, afr, false, b1,
                                                         (short)0, acc[1], false, false);
        acc[2] = __builtin_amdgcn_wmma_f32_16x16x32_bf16(false, afr, false, b2,
                                                         (short)0, acc[2], false, false);
        acc[3] = __builtin_amdgcn_wmma_f32_16x16x32_bf16(false, afr, false, b3,
                                                         (short)0, acc[3], false, false);
        __syncthreads();   // LDS buffer reuse safety before next DMA issue
    }

    float* outF          = (float*)OutV;
    unsigned short* outH = (unsigned short*)OutV;

    #pragma unroll
    for (int nt = 0; nt < 4; ++nt) {
        const int col = n0 + nt * 16 + laneLo;
        #pragma unroll
        for (int r = 0; r < 8; ++r) {
            const int row = m0 + laneHi * 8 + r;   // C/D: VGPR r -> M = r(+8)
            float v = acc[nt][r] * scale;
            if constexpr (EPI == EPI_F32_SCALE) {
                outF[ob + (long)row * ldo + col] = v;
            } else if constexpr (EPI == EPI_BF16) {
                if (bias) v += bias[col];
                outH[ob + (long)row * ldo + col] = f2bf(v);
            } else if constexpr (EPI == EPI_BF16_T) {
                if (bias) v += bias[col];
                outH[ob + (long)col * ldo + row] = f2bf(v);
            } else if constexpr (EPI == EPI_F32_BIAS_RES) {
                v += bias[col] + resid[(long)row * ldr + col];
                outF[ob + (long)row * ldo + col] = v;
            } else if constexpr (EPI == EPI_BF16_GELU) {
                v += bias[col];
                const float t = 0.7978845608028654f * (v + 0.044715f * v * v * v);
                outH[ob + (long)row * ldo + col] = f2bf(0.5f * v * (1.f + tanhf(t)));
            }
        }
    }
}

// ---------------------------------------------------------------------------
extern "C" void kernel_launch(void* const* d_in, const int* in_sizes, int n_in,
                              void* d_out, int out_size, void* d_ws, size_t ws_size,
                              hipStream_t stream) {
    (void)in_sizes; (void)n_in; (void)out_size; (void)ws_size;

    const float* x  = (const float*)d_in[0];
    const float* g1 = (const float*)d_in[1];
    const float* Wq = (const float*)d_in[2];
    const float* bq = (const float*)d_in[3];
    const float* Wk = (const float*)d_in[4];
    const float* bk = (const float*)d_in[5];
    const float* Wv = (const float*)d_in[6];
    const float* bv = (const float*)d_in[7];
    const float* Wo = (const float*)d_in[8];
    const float* bo = (const float*)d_in[9];
    const float* g2 = (const float*)d_in[10];
    const float* W1 = (const float*)d_in[11];
    const float* b1 = (const float*)d_in[12];
    const float* W2 = (const float*)d_in[13];
    const float* b2 = (const float*)d_in[14];
    float* out = (float*)d_out;

    constexpr int B = GB_B, T = GB_T, E = GB_E, H = GB_H, D = GB_D, I = GB_I;
    constexpr long BT = (long)B * T;

    char* base = (char*)d_ws;
    size_t off = 0;
    auto take = [&](size_t bytes) -> char* {
        char* r = base + off;
        off += (bytes + 255) & ~(size_t)255;
        return r;
    };
    unsigned short* WqT = (unsigned short*)take((size_t)E * E * 2);
    unsigned short* WkT = (unsigned short*)take((size_t)D * E * 2);
    unsigned short* WvT = (unsigned short*)take((size_t)D * E * 2);
    unsigned short* WoT = (unsigned short*)take((size_t)E * E * 2);
    unsigned short* W1T = (unsigned short*)take((size_t)I * E * 2);
    unsigned short* W2T = (unsigned short*)take((size_t)E * I * 2);
    unsigned short* hB  = (unsigned short*)take((size_t)BT * E * 2);
    unsigned short* qB  = (unsigned short*)take((size_t)BT * E * 2);
    unsigned short* kB  = (unsigned short*)take((size_t)BT * D * 2);
    unsigned short* vT  = (unsigned short*)take((size_t)B * D * T * 2);
    float*          sc  = (float*)take((size_t)B * H * T * T * 4);
    unsigned short* pB  = (unsigned short*)take((size_t)B * H * T * T * 2);
    unsigned short* oB  = (unsigned short*)take((size_t)BT * E * 2);
    float*          att = (float*)take((size_t)BT * E * 4);
    unsigned short* fB  = (unsigned short*)take((size_t)BT * E * 2);
    unsigned short* mB  = (unsigned short*)take((size_t)BT * I * 2);

    const dim3 tb(32, 8);
    // -- weight convert + transpose (fp32 [K,N] -> bf16 [N,K]) --
    transpose_f32_bf16<<<dim3(E / 32, E / 32), tb, 0, stream>>>(Wq, WqT, E, E);
    transpose_f32_bf16<<<dim3(D / 32, E / 32), tb, 0, stream>>>(Wk, WkT, E, D);
    transpose_f32_bf16<<<dim3(D / 32, E / 32), tb, 0, stream>>>(Wv, WvT, E, D);
    transpose_f32_bf16<<<dim3(E / 32, E / 32), tb, 0, stream>>>(Wo, WoT, E, E);
    transpose_f32_bf16<<<dim3(I / 32, E / 32), tb, 0, stream>>>(W1, W1T, E, I);
    transpose_f32_bf16<<<dim3(E / 32, I / 32), tb, 0, stream>>>(W2, W2T, I, E);

    // -- h = rmsnorm(x, g1) --
    rmsnorm_bf16<<<(int)BT, 256, 0, stream>>>(x, g1, hB, E);

    // -- q/k projections --
    gemm_bf16_wmma<EPI_BF16><<<dim3(E / 64, BT / 128, 1), 256, 0, stream>>>(
        hB, E, 0, 0, WqT, E, 0, 0, qB, E, 0, 0, 1, bq, nullptr, 0, E, 1.f);
    gemm_bf16_wmma<EPI_BF16><<<dim3(D / 64, BT / 128, 1), 256, 0, stream>>>(
        hB, E, 0, 0, WkT, E, 0, 0, kB, D, 0, 0, 1, bk, nullptr, 0, E, 1.f);
    // -- v projection, stored transposed per batch: vT[b] = (h[b] @ Wv)^T [D,T] --
    gemm_bf16_wmma<EPI_BF16_T><<<dim3(D / 64, T / 128, B), 256, 0, stream>>>(
        hB, E, (long)T * E, 0, WvT, E, 0, 0, vT, T, (long)D * T, 0,
        1, bv, nullptr, 0, E, 1.f);

    // -- scores[b,h] = q[b,h] @ k[b]^T * sqrt(D)  (faithful quirk: *16) --
    gemm_bf16_wmma<EPI_F32_SCALE><<<dim3(T / 64, T / 128, B * H), 256, 0, stream>>>(
        qB, E, (long)T * E, (long)D,
        kB, D, (long)T * D, 0,
        sc, T, (long)H * T * T, (long)T * T,
        H, nullptr, nullptr, 0, D, 16.f);

    // -- probs = softmax(scores) -> bf16 --
    softmax_bf16<<<B * H * T, 256, 0, stream>>>(sc, pB, T);

    // -- o[b,h] = probs[b,h] @ v[b]  (Bt = vT[b]), scattered into [B,T,E] --
    gemm_bf16_wmma<EPI_BF16><<<dim3(D / 64, T / 128, B * H), 256, 0, stream>>>(
        pB, T, (long)H * T * T, (long)T * T,
        vT, T, (long)D * T, 0,
        oB, E, (long)T * E, (long)D,
        H, nullptr, nullptr, 0, T, 1.f);

    // -- attn_out = o @ Wo + bo + x --
    gemm_bf16_wmma<EPI_F32_BIAS_RES><<<dim3(E / 64, BT / 128, 1), 256, 0, stream>>>(
        oB, E, 0, 0, WoT, E, 0, 0, att, E, 0, 0, 1, bo, x, E, E, 1.f);

    // -- f = rmsnorm(attn_out, g2) --
    rmsnorm_bf16<<<(int)BT, 256, 0, stream>>>(att, g2, fB, E);

    // -- m = gelu(f @ W1 + b1) --
    gemm_bf16_wmma<EPI_BF16_GELU><<<dim3(I / 64, BT / 128, 1), 256, 0, stream>>>(
        fB, E, 0, 0, W1T, E, 0, 0, mB, I, 0, 0, 1, b1, nullptr, 0, E, 1.f);

    // -- out = m @ W2 + b2 + attn_out --
    gemm_bf16_wmma<EPI_F32_BIAS_RES><<<dim3(E / 64, BT / 128, 1), 256, 0, stream>>>(
        mB, I, 0, 0, W2T, I, 0, 0, out, E, 0, 0, 1, b2, att, E, I, 1.f);
}
